// AttentionModule_53798760349756
// MI455X (gfx1250) — matrix operand order repaired and checked by reading, not compiled, not run on hardware
//
#include <hip/hip_runtime.h>
#include <hip/hip_bf16.h>

// ---------------- problem constants (match reference) ----------------
#define D_MODEL 2048
#define SEQ_T   2048
#define BATCH   2
#define NHEADS  32
#define NKV     8
#define HDIM    64
#define KVDIM   (NKV * HDIM)     // 512
#define BT      (BATCH * SEQ_T)  // 4096

// ---------------- WMMA types (gfx1250, wave32) ----------------
typedef __bf16 bf16x16 __attribute__((ext_vector_type(16)));
typedef float  f32x8   __attribute__((ext_vector_type(8)));

struct alignas(16) U128 { unsigned int w[4]; };
union FragU { U128 u[2]; bf16x16 v; };

__device__ __forceinline__ unsigned short f2bf(float f) {
  unsigned u = __builtin_bit_cast(unsigned, f);
  u += 0x7fffu + ((u >> 16) & 1u);          // round-to-nearest-even
  return (unsigned short)(u >> 16);
}
__device__ __forceinline__ float bf2f(unsigned short s) {
  unsigned u = ((unsigned)s) << 16;
  return __builtin_bit_cast(float, u);
}

// A fragment: 16x32 bf16 (MxK), row-major source with leading dim ld.
// ISA layout: lanes 0-15 row M=r hold K = 0..7 (V0..V3) and 16..23 (V4..V7);
// lanes 16-31 hold K = 8..15 and 24..31.  -> two contiguous 16B chunks/lane.
__device__ __forceinline__ bf16x16 load_frag_a(const unsigned short* __restrict__ A,
                                               int ld, int m0, int k0, int lane) {
  int r = lane & 15, h = lane >> 4;
  const unsigned short* p = A + (size_t)(m0 + r) * ld + k0 + h * 8;
  FragU f;
  f.u[0] = *(const U128*)(p);
  f.u[1] = *(const U128*)(p + 16);
  return f.v;
}

// B fragment: 32x16 bf16 (KxN), with B[k,n] = S[n0+n, k0+k] (row-major src, ld).
// ISA layout: lanes 0-15 col N=n hold K=0..7 and 16..23, lanes 16-31 hold K=8..15 and 24..31
// -> two contiguous 16B chunks per lane.
__device__ __forceinline__ bf16x16 load_frag_b(const unsigned short* __restrict__ S,
                                               int ld, int n0, int k0, int lane) {
  int n = lane & 15, h = lane >> 4;
  const unsigned short* p = S + (size_t)(n0 + n) * ld + k0 + h * 8;
  FragU f;
  f.u[0] = *(const U128*)(p);
  f.u[1] = *(const U128*)(p + 16);
  return f.v;
}

// ---------------- fp32 -> bf16 conversion (vectorized) ----------------
__global__ void f32_to_bf16_kernel(const float* __restrict__ in,
                                   unsigned short* __restrict__ out, int n4) {
  int i = blockIdx.x * blockDim.x + threadIdx.x;
  if (i >= n4) return;
  float4 f = ((const float4*)in)[i];
  unsigned lo = (unsigned)f2bf(f.x) | ((unsigned)f2bf(f.y) << 16);
  unsigned hi = (unsigned)f2bf(f.z) | ((unsigned)f2bf(f.w) << 16);
  ((unsigned int*)out)[2 * i + 0] = lo;
  ((unsigned int*)out)[2 * i + 1] = hi;
}

// ---------------- GEMM: C[M,N] = A[M,Kd] * W[N,Kd]^T  (bf16 in, f32 acc) ----
// Block = 256 threads (8 waves) owning a 128x64 C macro-tile.
// B tile (64 N-rows x 64 K) is staged in LDS with CDNA5 async global->LDS
// copies (ASYNCcnt), double-buffered; waves read fragments via ds_load_b128.
template <bool OUT_BF16>
__global__ void __launch_bounds__(256)
gemm_xwt(const unsigned short* __restrict__ A,
         const unsigned short* __restrict__ W,
         void* __restrict__ C, int M, int N, int Kd) {
  __shared__ alignas(16) unsigned short Bl[2][64 * 64];  // 2 x 8KB
  int tid  = threadIdx.x;
  int lane = tid & 31;
  int wave = tid >> 5;
  int m0 = (blockIdx.y * 8 + wave) * 16;
  int n0 = blockIdx.x * 64;

  // Cooperative copy mapping: 512 x 16B chunks, 2 consecutive chunks/thread.
  int c0 = tid * 2;
  int bi = c0 >> 3;                       // tile row (W row = n0 + bi)
  int bj = c0 & 7;                        // 16B chunk within the 128B row
  unsigned lds_base = (unsigned)(size_t)(&Bl[0][0]);  // low 32 bits = LDS offset
  unsigned lds_off0 = (unsigned)(bi * 128 + bj * 16);
  unsigned grow     = (unsigned)((n0 + bi) * Kd + bj * 8) * 2u;  // byte offset

  auto issue_copy = [&](int k0, int buf) {
    unsigned l0 = lds_base + (unsigned)buf * (64u * 64u * 2u) + lds_off0;
    unsigned g0 = grow + (unsigned)k0 * 2u;
    asm volatile("global_load_async_to_lds_b128 %0, %1, %2"
                 :: "v"(l0), "v"(g0), "s"(W) : "memory");
    asm volatile("global_load_async_to_lds_b128 %0, %1, %2"
                 :: "v"(l0 + 16u), "v"(g0 + 16u), "s"(W) : "memory");
  };

  f32x8 acc[4] = {};
  int nsteps = Kd >> 6;                   // K-step = 64
  issue_copy(0, 0);
  for (int s = 0; s < nsteps; ++s) {
    int k0 = s << 6;
    if (s + 1 < nsteps) {
      issue_copy(k0 + 64, (s + 1) & 1);
      // async loads complete in order: <=2 outstanding => tile s has landed
      asm volatile("s_wait_asynccnt 0x2" ::: "memory");
    } else {
      asm volatile("s_wait_asynccnt 0x0" ::: "memory");
    }
    __syncthreads();

    const unsigned short* Bcur = &Bl[s & 1][0];
    bf16x16 a0 = load_frag_a(A, Kd, m0, k0, lane);
    bf16x16 a1 = load_frag_a(A, Kd, m0, k0 + 32, lane);
#pragma unroll
    for (int t = 0; t < 4; ++t) {
      bf16x16 b0 = load_frag_b(Bcur, 64, t * 16, 0, lane);   // ds_load_b128 x2
      acc[t] = __builtin_amdgcn_wmma_f32_16x16x32_bf16(
          false, a0, false, b0, (short)0, acc[t], false, false);
      bf16x16 b1 = load_frag_b(Bcur, 64, t * 16, 32, lane);
      acc[t] = __builtin_amdgcn_wmma_f32_16x16x32_bf16(
          false, a1, false, b1, (short)0, acc[t], false, false);
    }
    __syncthreads();  // all waves done reading before buffer reuse
  }

  // C layout: lanes 0-15: VGPR j -> M=j; lanes 16-31: VGPR j -> M=8+j; N = lane&15
  int h = lane >> 4, nn = lane & 15;
#pragma unroll
  for (int t = 0; t < 4; ++t)
#pragma unroll
    for (int j = 0; j < 8; ++j) {
      size_t idx = (size_t)(m0 + h * 8 + j) * N + (n0 + t * 16 + nn);
      if constexpr (OUT_BF16)
        ((unsigned short*)C)[idx] = f2bf(acc[t][j]);
      else
        ((float*)C)[idx] = acc[t][j];
    }
}

// ---------------- RoPE in-place on bf16 Q [BT,2048] and K [BT,512] ----------
__global__ void __launch_bounds__(1024)
rope_kernel(unsigned short* __restrict__ q, unsigned short* __restrict__ k) {
  int row = blockIdx.x;                 // b*T + t
  int t = row & (SEQ_T - 1);
  int h = threadIdx.x >> 5, i = threadIdx.x & 31;  // head, freq index
  float invf = __powf(10000.0f, -(float)i * (1.0f / 32.0f));
  float sn, cs;
  __sincosf((float)t * invf, &sn, &cs);
  unsigned short* qp = q + (size_t)row * D_MODEL + h * HDIM;
  float q0 = bf2f(qp[i]), q1 = bf2f(qp[i + 32]);
  qp[i]      = f2bf(q0 * cs - q1 * sn);
  qp[i + 32] = f2bf(q1 * cs + q0 * sn);
  if (h < NKV) {
    unsigned short* kp = k + (size_t)row * KVDIM + h * HDIM;
    float k0 = bf2f(kp[i]), k1 = bf2f(kp[i + 32]);
    kp[i]      = f2bf(k0 * cs - k1 * sn);
    kp[i + 32] = f2bf(k1 * cs + k0 * sn);
  }
}

// ---------------- V transpose: [b*T, kvh*64+d] -> [(b*8+kvh)*64+d, T] --------
__global__ void vtrans_kernel(const unsigned short* __restrict__ v,
                              unsigned short* __restrict__ vt) {
  int idx = blockIdx.x * blockDim.x + threadIdx.x;
  if (idx >= BT * KVDIM) return;
  int row = idx / KVDIM;
  int col = idx - row * KVDIM;
  int b = row / SEQ_T, t = row - b * SEQ_T;
  vt[((size_t)b * KVDIM + col) * SEQ_T + t] = v[idx];
}

// ---------------- Flash attention: one wave per (q-tile, b, h) --------------
__global__ void __launch_bounds__(32)
flash_kernel(const unsigned short* __restrict__ qb,
             const unsigned short* __restrict__ kb,
             const unsigned short* __restrict__ vt,
             unsigned short* __restrict__ ab) {
  __shared__ alignas(16) unsigned short Plds[16][32];  // P: C-layout f32 -> A-layout bf16
  int lane = threadIdx.x;
  int qt = blockIdx.x;                 // query tile (16 rows)
  int bh = blockIdx.y;
  int b = bh >> 5, h = bh & 31;
  int kvh = h >> 2;                    // GQA: 4 query heads per KV head
  int g = lane >> 4, nn = lane & 15;

  // Q fragments (held for the whole key loop)
  bf16x16 qa0 = load_frag_a(qb, D_MODEL, b * SEQ_T + qt * 16, h * HDIM, lane);
  bf16x16 qa1 = load_frag_a(qb, D_MODEL, b * SEQ_T + qt * 16, h * HDIM + 32, lane);

  f32x8 oa[4] = {};                    // O accumulator 16x64 in C layout
  float runmax[8], runsum[8];
#pragma unroll
  for (int j = 0; j < 8; ++j) { runmax[j] = -1e30f; runsum[j] = 0.f; }

  int ktEnd = (qt * 16 + 15) >> 5;     // inclusive, causal
  for (int kt = 0; kt <= ktEnd; ++kt) {
    // S = Q @ K^T over 32 keys (two 16-key column halves)
    f32x8 s0 = {}, s1 = {};
    bf16x16 kf;
    kf = load_frag_b(kb, KVDIM, b * SEQ_T + kt * 32, kvh * HDIM, lane);
    s0 = __builtin_amdgcn_wmma_f32_16x16x32_bf16(false, qa0, false, kf, (short)0, s0, false, false);
    kf = load_frag_b(kb, KVDIM, b * SEQ_T + kt * 32, kvh * HDIM + 32, lane);
    s0 = __builtin_amdgcn_wmma_f32_16x16x32_bf16(false, qa1, false, kf, (short)0, s0, false, false);
    kf = load_frag_b(kb, KVDIM, b * SEQ_T + kt * 32 + 16, kvh * HDIM, lane);
    s1 = __builtin_amdgcn_wmma_f32_16x16x32_bf16(false, qa0, false, kf, (short)0, s1, false, false);
    kf = load_frag_b(kb, KVDIM, b * SEQ_T + kt * 32 + 16, kvh * HDIM + 32, lane);
    s1 = __builtin_amdgcn_wmma_f32_16x16x32_bf16(false, qa1, false, kf, (short)0, s1, false, false);

    const float scale = 0.125f;        // 1/sqrt(64)
    float p0[8], p1[8], fac[8];
#pragma unroll
    for (int j = 0; j < 8; ++j) {
      int qg = qt * 16 + g * 8 + j;    // global query row for this element
      float v0 = s0[j] * scale; if (kt * 32 + nn > qg)      v0 = -1e30f;
      float v1 = s1[j] * scale; if (kt * 32 + 16 + nn > qg) v1 = -1e30f;
      // row-max across the 16 lanes that hold this row (xor within 16-group)
      float tm = fmaxf(v0, v1);
      tm = fmaxf(tm, __shfl_xor(tm, 1, 32));
      tm = fmaxf(tm, __shfl_xor(tm, 2, 32));
      tm = fmaxf(tm, __shfl_xor(tm, 4, 32));
      tm = fmaxf(tm, __shfl_xor(tm, 8, 32));
      float nm = fmaxf(runmax[j], tm);
      fac[j] = __expf(runmax[j] - nm);
      runmax[j] = nm;
      p0[j] = __expf(v0 - nm);
      p1[j] = __expf(v1 - nm);
      float ps = p0[j] + p1[j];
      ps += __shfl_xor(ps, 1, 32);
      ps += __shfl_xor(ps, 2, 32);
      ps += __shfl_xor(ps, 4, 32);
      ps += __shfl_xor(ps, 8, 32);
      runsum[j] = runsum[j] * fac[j] + ps;
    }
    // rescale output accumulators (same row<->element mapping as S)
#pragma unroll
    for (int t = 0; t < 4; ++t)
#pragma unroll
      for (int j = 0; j < 8; ++j) oa[t][j] *= fac[j];

    // stage P (C layout) into LDS as bf16, reload in A layout
#pragma unroll
    for (int j = 0; j < 8; ++j) {
      Plds[g * 8 + j][nn]      = f2bf(p0[j]);
      Plds[g * 8 + j][16 + nn] = f2bf(p1[j]);
    }
    __syncthreads();
    FragU pf;
    pf.u[0] = *(const U128*)(&Plds[nn][g * 8]);
    pf.u[1] = *(const U128*)(&Plds[nn][16 + g * 8]);
    bf16x16 pa = pf.v;
    __syncthreads();

    // O += P @ V ; V^T fragment: B[k=key, n=d] contiguous along keys
#pragma unroll
    for (int t = 0; t < 4; ++t) {
      const unsigned short* vp =
          vt + ((size_t)b * KVDIM + kvh * HDIM + t * 16 + nn) * SEQ_T + kt * 32 + g * 8;
      FragU vf;
      vf.u[0] = *(const U128*)(vp);
      vf.u[1] = *(const U128*)(vp + 16);
      oa[t] = __builtin_amdgcn_wmma_f32_16x16x32_bf16(
          false, pa, false, vf.v, (short)0, oa[t], false, false);
    }
  }

  // normalize and store attention output (bf16, [b*T+t, h*64+d])
#pragma unroll
  for (int t = 0; t < 4; ++t)
#pragma unroll
    for (int j = 0; j < 8; ++j) {
      float o = oa[t][j] / runsum[j];
      size_t idx = (size_t)(b * SEQ_T + qt * 16 + g * 8 + j) * D_MODEL +
                   h * HDIM + t * 16 + nn;
      ab[idx] = f2bf(o);
    }
}

// ---------------- host-side launch ----------------
extern "C" void kernel_launch(void* const* d_in, const int* in_sizes, int n_in,
                              void* d_out, int out_size, void* d_ws, size_t ws_size,
                              hipStream_t stream) {
  (void)in_sizes; (void)n_in; (void)out_size; (void)ws_size;
  const float* x  = (const float*)d_in[0];
  const float* Wq = (const float*)d_in[1];
  const float* Wk = (const float*)d_in[2];
  const float* Wv = (const float*)d_in[3];
  const float* Wo = (const float*)d_in[4];
  float* out = (float*)d_out;

  unsigned short* ws = (unsigned short*)d_ws;
  size_t o = 0;
  unsigned short* xb  = ws + o; o += (size_t)BT * D_MODEL;       // reused as attn-out
  unsigned short* wqb = ws + o; o += (size_t)D_MODEL * D_MODEL;
  unsigned short* wkb = ws + o; o += (size_t)KVDIM * D_MODEL;
  unsigned short* wvb = ws + o; o += (size_t)KVDIM * D_MODEL;
  unsigned short* wob = ws + o; o += (size_t)D_MODEL * D_MODEL;
  unsigned short* qbp = ws + o; o += (size_t)BT * D_MODEL;
  unsigned short* kbp = ws + o; o += (size_t)BT * KVDIM;
  unsigned short* vbp = ws + o; o += (size_t)BT * KVDIM;
  unsigned short* vtp = ws + o; o += (size_t)BT * KVDIM;
  // total: 33,554,432 bf16 = 64 MiB of workspace

  auto conv = [&](const float* src, unsigned short* dst, size_t n) {
    int n4 = (int)(n / 4);
    f32_to_bf16_kernel<<<(n4 + 255) / 256, 256, 0, stream>>>(src, dst, n4);
  };
  conv(x,  xb,  (size_t)BT * D_MODEL);
  conv(Wq, wqb, (size_t)D_MODEL * D_MODEL);
  conv(Wk, wkb, (size_t)KVDIM * D_MODEL);
  conv(Wv, wvb, (size_t)KVDIM * D_MODEL);
  conv(Wo, wob, (size_t)D_MODEL * D_MODEL);

  // QKV projections (bf16 out); block = 256 threads owning 128x64 C tiles
  gemm_xwt<true><<<dim3(D_MODEL / 64, BT / 128), 256, 0, stream>>>(
      xb, wqb, qbp, BT, D_MODEL, D_MODEL);
  gemm_xwt<true><<<dim3(KVDIM / 64, BT / 128), 256, 0, stream>>>(
      xb, wkb, kbp, BT, KVDIM, D_MODEL);
  gemm_xwt<true><<<dim3(KVDIM / 64, BT / 128), 256, 0, stream>>>(
      xb, wvb, vbp, BT, KVDIM, D_MODEL);

  rope_kernel<<<BT, 1024, 0, stream>>>(qbp, kbp);
  vtrans_kernel<<<(BT * KVDIM) / 256, 256, 0, stream>>>(vbp, vtp);

  unsigned short* abp = xb;  // x_bf16 dead after QKV GEMMs -> reuse for attn out
  flash_kernel<<<dim3(SEQ_T / 16, BATCH * NHEADS), 32, 0, stream>>>(
      qbp, kbp, vtp, abp);

  // output projection (f32 out)
  gemm_xwt<false><<<dim3(D_MODEL / 64, BT / 128), 256, 0, stream>>>(
      abp, wob, out, BT, D_MODEL, D_MODEL);
}
